// RNNModel_3169685865039
// MI455X (gfx1250) — matrix operand order repaired and checked by reading, not compiled, not run on hardware
//
#include <hip/hip_runtime.h>
#include <cstdint>

#define NHID 1024
#define NTOK 10000
#define TT   128
#define BB   32
#define SPLITK 4

typedef __attribute__((ext_vector_type(16))) __bf16         v16bf;
typedef __attribute__((ext_vector_type(16))) unsigned short v16u;
typedef __attribute__((ext_vector_type(8)))  float          v8f;

// ---------------------------------------------------------------- helpers
__device__ __forceinline__ unsigned short f2bf(float f) {
  unsigned int u = __float_as_uint(f);
  u += 0x7FFFu + ((u >> 16) & 1u);          // round-to-nearest-even
  return (unsigned short)(u >> 16);
}

// Load one pre-packed 512-element bf16 fragment (A 16x32 or B 32x16):
// fragment index = tile*ktiles + kt, lane-contiguous 32B -> two global_load_b128.
__device__ __forceinline__ v16bf load_frag(const unsigned short* P, int ktiles,
                                           int tile, int kt, int lane) {
  const unsigned short* p = P + (((size_t)tile * ktiles + kt) << 9) + lane * 16;
  v16u r = *(const v16u*)p;
  return __builtin_bit_cast(v16bf, r);
}

// Scatter one element into packed A-fragment layout.
// 16-bit A 16x32 layout: lanes 0-15 hold K {0..7,16..23}; lanes 16-31 K {8..15,24..31}.
__device__ __forceinline__ void store_a_elem(unsigned short* ap, int ktiles,
                                             int mrow, int k, unsigned short v) {
  int mt = mrow >> 4, mr = mrow & 15;
  int kt = k >> 5,   kk = k & 31;
  int hi = (kk >> 3) & 1;                       // which half-wave holds it
  int j  = (kk & 7) + ((kk >> 4) << 3);         // kk>=16 -> upper 8 halfs
  int lane = mr + hi * 16;
  ap[(((size_t)mt * ktiles + kt) << 9) + lane * 16 + j] = v;
}

// C/D 16x16 f32 store: lanes0-15: N=lane, M=r; lanes16-31: N=lane-16, M=8+r.
__device__ __forceinline__ void store_tile(float* out, int ld, int m_base,
                                           int n_base, v8f acc, int lane) {
  int n = n_base + (lane & 15);
  int mrow = m_base + ((lane >> 4) & 1) * 8;
#pragma unroll
  for (int r = 0; r < 8; ++r) out[(size_t)(mrow + r) * ld + n] = acc[r];
}

// ---------------------------------------------------------------- pack weights
// Pre-pack f32 weights into bf16 WMMA B-fragment order (one-time).
// Dense 16x16x32 B layout: lanes0-15: N=lane, K={0..7,16..23}; lanes16-31: N=lane-16, K={8..15,24..31}.
__global__ void __launch_bounds__(256)
pack_w(const float* __restrict__ src, unsigned short* __restrict__ dst,
       int ktiles, int ntiles, int srcld, int transpose) {
  size_t idx = (size_t)blockIdx.x * 256 + threadIdx.x;
  size_t total = (size_t)ktiles * ntiles * 512;
  if (idx >= total) return;
  int j    = (int)(idx & 15);
  int lane = (int)((idx >> 4) & 31);
  size_t f = idx >> 9;
  int kt = (int)(f % ktiles);
  int nt = (int)(f / ktiles);
  int n  = nt * 16 + (lane & 15);
  int kk = kt * 32 + ((lane >> 4) & 1) * 8 + (j & 7) + (j >> 3) * 16;
  size_t si = transpose ? ((size_t)n * srcld + kk) : ((size_t)kk * srcld + n);
  dst[idx] = f2bf(src[si]);
}

// ---------------------------------------------------------------- A packing
// Per step: pack emb half of xh (k = 0..1023), fused embedding gather.
__global__ void __launch_bounds__(256)
pack_x(const float* __restrict__ encW, const int* __restrict__ tok,
       unsigned short* __restrict__ xhPk) {
  int idx = blockIdx.x * 256 + threadIdx.x;     // 0 .. 32*1024-1
  int mrow = idx >> 10, k = idx & (NHID - 1);
  float v = encW[(size_t)tok[mrow] * NHID + k];
  store_a_elem(xhPk, 64, mrow, k, f2bf(v));
}

// Once at t=0: pack h0 into the h half of xh (k = 1024..2047).
__global__ void __launch_bounds__(256)
pack_h0(const float* __restrict__ h, unsigned short* __restrict__ xhPk) {
  int idx = blockIdx.x * 256 + threadIdx.x;
  int mrow = idx >> 10, n = idx & (NHID - 1);
  store_a_elem(xhPk, 64, mrow, NHID + n, f2bf(h[idx]));
}

// ---------------------------------------------------------------- cell phases
struct PhaseArgs {
  const unsigned short* W0a; const unsigned short* W1a; const unsigned short* W2a;
  const unsigned short* pa0; const unsigned short* pa1; const unsigned short* pa2;
  const float* s0a; const float* s1a; const float* s2a;   // sp (f32, for epilogue)
  float* d0a; float* d1a; float* d2a;                     // s_out f32
  unsigned short* pd0; unsigned short* pd1; unsigned short* pd2; // s_out packed bf16
  int act0, act1, act2;          // 0 sigmoid, 1 relu, 2 tanh, 3 identity
  int ktiles;                    // K/32
};

// grid: (16 n-blocks of 64 cols, SPLITK, nEdges), block 256 (8 waves).
// wave w: m_tile = w&1, n_sub = w>>1; computes c-tile (n) and h-tile (n+1024).
// Inner loop: pure b128 fragment loads + 2 WMMAs.
__global__ void __launch_bounds__(256)
rnn_gemm(PhaseArgs ph, float* __restrict__ partial) {
  int z = blockIdx.z;
  const unsigned short* W  = (z == 0) ? ph.W0a : (z == 1) ? ph.W1a : ph.W2a;
  const unsigned short* PA = (z == 0) ? ph.pa0 : (z == 1) ? ph.pa1 : ph.pa2;
  int lane = threadIdx.x & 31;
  int wv   = threadIdx.x >> 5;
  int mtile = wv & 1;
  int n_c   = blockIdx.x * 64 + (wv >> 1) * 16;
  int ktiles = ph.ktiles;
  int kper   = ktiles / SPLITK;
  int kt0    = blockIdx.y * kper;
  int ntc = n_c >> 4;
  int nth = (n_c + NHID) >> 4;
  v8f acc_c = {}; v8f acc_h = {};
#pragma unroll 4
  for (int kt = kt0; kt < kt0 + kper; ++kt) {
    v16bf a  = load_frag(PA, ktiles, mtile, kt, lane);
    v16bf bc = load_frag(W,  ktiles, ntc,   kt, lane);
    v16bf bh = load_frag(W,  ktiles, nth,   kt, lane);
    acc_c = __builtin_amdgcn_wmma_f32_16x16x32_bf16(false, a, false, bc,
                                                    (short)0, acc_c, false, false);
    acc_h = __builtin_amdgcn_wmma_f32_16x16x32_bf16(false, a, false, bh,
                                                    (short)0, acc_h, false, false);
  }
  float* pz = partial + ((size_t)z * SPLITK + blockIdx.y) * (BB * 2048);
  store_tile(pz, 2048, mtile << 4, n_c, acc_c, lane);
  store_tile(pz, 2048, mtile << 4, n_c + NHID, acc_h, lane);
}

// split-K reduce + DARTS update: s = sp + sigmoid(c)*(act(h)-sp);
// writes s both as f32 (next epilogues) and packed bf16 (next GEMM A operand).
__global__ void __launch_bounds__(256)
rnn_epi(PhaseArgs ph, const float* __restrict__ partial) {
  int z = blockIdx.z;
  const float* src   = (z == 0) ? ph.s0a : (z == 1) ? ph.s1a : ph.s2a;
  float* dst         = (z == 0) ? ph.d0a : (z == 1) ? ph.d1a : ph.d2a;
  unsigned short* pd = (z == 0) ? ph.pd0 : (z == 1) ? ph.pd1 : ph.pd2;
  int act            = (z == 0) ? ph.act0 : (z == 1) ? ph.act1 : ph.act2;
  int idx = blockIdx.x * 256 + threadIdx.x;             // 0 .. 32*1024-1
  int mrow = idx >> 10, n = idx & (NHID - 1);
  const float* pz = partial + (size_t)z * SPLITK * (BB * 2048);
  float c = 0.f, h = 0.f;
#pragma unroll
  for (int kp = 0; kp < SPLITK; ++kp) {
    c += pz[kp * (BB * 2048) + mrow * 2048 + n];
    h += pz[kp * (BB * 2048) + mrow * 2048 + NHID + n];
  }
  float sp = src[(size_t)mrow * NHID + n];
  float g = 1.f / (1.f + __expf(-c));
  float hv;
  if (act == 0)      hv = 1.f / (1.f + __expf(-h));
  else if (act == 1) hv = fmaxf(h, 0.f);
  else if (act == 2) hv = tanhf(h);
  else               hv = h;
  float sv = sp + g * (hv - sp);
  dst[idx] = sv;
  store_a_elem(pd, 32, mrow, n, f2bf(sv));
}

// h_new = mean(s1..s8); write f32 h (for next phase-0 epilogue + h_last),
// packed bf16 into xh (next step's W0 A operand) and into hiddens (decoder A).
__global__ void __launch_bounds__(256)
finalize_h(const float* __restrict__ S, float* __restrict__ hbuf,
           unsigned short* __restrict__ xhPk, unsigned short* __restrict__ hidPk,
           int t) {
  int idx = blockIdx.x * 256 + threadIdx.x;
  int mrow = idx >> 10, n = idx & (NHID - 1);
  float a = 0.f;
#pragma unroll
  for (int i = 1; i <= 8; ++i) a += S[i * (BB * NHID) + idx];
  float v = a * 0.125f;
  hbuf[idx] = v;
  unsigned short b = f2bf(v);
  store_a_elem(xhPk, 64, mrow, NHID + n, b);
  store_a_elem(hidPk, 32, t * BB + mrow, n, b);
}

// ---------------------------------------------------------------- decoder
// logits[r][n] = hiddens[r] . encW[n]; both operands pre-packed bf16 fragments.
// grid (625 n-tiles, 64 row-blocks of 64), 4 waves.
__global__ void __launch_bounds__(128)
dec_gemm(const unsigned short* __restrict__ hidPk,
         const unsigned short* __restrict__ Wd, float* __restrict__ out) {
  int lane = threadIdx.x & 31;
  int wv   = threadIdx.x >> 5;
  int rowb = blockIdx.y * 64 + wv * 16;
  int mtile = rowb >> 4;
  int nt   = blockIdx.x;
  v8f acc = {};
#pragma unroll 8
  for (int kt = 0; kt < NHID / 32; ++kt) {
    if (kt + 4 < NHID / 32)   // stream B ahead into cache (global_prefetch_b8)
      __builtin_prefetch(Wd + (((size_t)nt * (NHID / 32) + kt + 4) << 9) + lane * 16, 0, 1);
    v16bf a = load_frag(hidPk, NHID / 32, mtile, kt, lane);
    v16bf b = load_frag(Wd,    NHID / 32, nt,    kt, lane);
    acc = __builtin_amdgcn_wmma_f32_16x16x32_bf16(false, a, false, b,
                                                  (short)0, acc, false, false);
  }
  store_tile(out, NTOK, rowb, nt * 16, acc, lane);
}

// in-place log_softmax(logits + bias) per row of 10000
__global__ void __launch_bounds__(256)
logsoftmax(float* __restrict__ out, const float* __restrict__ bias) {
  __shared__ float red[256];
  int r = blockIdx.x, tid = threadIdx.x;
  float* row = out + (size_t)r * NTOK;
  float mx = -3.4e38f;
  for (int v = tid; v < NTOK; v += 256) mx = fmaxf(mx, row[v] + bias[v]);
  red[tid] = mx; __syncthreads();
  for (int s = 128; s > 0; s >>= 1) {
    if (tid < s) red[tid] = fmaxf(red[tid], red[tid + s]);
    __syncthreads();
  }
  float rowmax = red[0]; __syncthreads();
  float sum = 0.f;
  for (int v = tid; v < NTOK; v += 256) sum += __expf(row[v] + bias[v] - rowmax);
  red[tid] = sum; __syncthreads();
  for (int s = 128; s > 0; s >>= 1) {
    if (tid < s) red[tid] += red[tid + s];
    __syncthreads();
  }
  float lse = __logf(red[0]) + rowmax;
  for (int v = tid; v < NTOK; v += 256) row[v] = row[v] + bias[v] - lse;
}

// ---------------------------------------------------------------- launch
extern "C" void kernel_launch(void* const* d_in, const int* in_sizes, int n_in,
                              void* d_out, int out_size, void* d_ws, size_t ws_size,
                              hipStream_t stream) {
  const int*   tokens = (const int*)d_in[0];     // (128,32)
  const float* h0     = (const float*)d_in[1];   // (32,1024)
  const float* encW   = (const float*)d_in[2];   // (10000,1024)
  const float* W0     = (const float*)d_in[3];   // (2048,2048)
  const float* Ws     = (const float*)d_in[4];   // (8,1024,2048)
  const float* decb   = (const float*)d_in[5];   // (10000,)
  float* out = (float*)d_out;

  char* ws = (char*)d_ws;
  size_t off = 0;
  auto take = [&](size_t n) -> char* {
    char* p = ws + off; off = (off + n + 255) & ~(size_t)255; return p;
  };
  unsigned short* W0p  = (unsigned short*)take(2048ull * 2048 * 2);
  unsigned short* Wsp  = (unsigned short*)take(8ull * 1024 * 2048 * 2);
  unsigned short* Dp   = (unsigned short*)take((size_t)NHID * NTOK * 2);
  float* S    = (float*)take(9ull * BB * NHID * 4);
  float* hbuf = (float*)take((size_t)BB * NHID * 4);
  float* part = (float*)take((size_t)3 * SPLITK * BB * 2048 * 4);
  unsigned short* APK  = (unsigned short*)take(9ull * BB * NHID * 2);   // s0..s8 packed
  unsigned short* xhPk = (unsigned short*)take((size_t)BB * 2048 * 2);  // [emb|h] packed
  unsigned short* hidPk = (unsigned short*)take((size_t)TT * BB * NHID * 2);

  // one-time bf16 fragment packing (L2-resident afterwards)
  pack_w<<<(2048 * 2048 + 255) / 256, 256, 0, stream>>>(W0, W0p, 64, 128, 2048, 0);
  for (int i = 0; i < 8; ++i)
    pack_w<<<(1024 * 2048 + 255) / 256, 256, 0, stream>>>(
        Ws + (size_t)i * 1024 * 2048, Wsp + (size_t)i * 1024 * 2048, 32, 128, 2048, 0);
  pack_w<<<(NHID * NTOK + 255) / 256, 256, 0, stream>>>(encW, Dp, 32, 625, NHID, 1);
  pack_h0<<<(BB * NHID) / 256, 256, 0, stream>>>(h0, xhPk);

  float* s[9];
  unsigned short* ap[9];
  for (int i = 0; i < 9; ++i) { s[i] = S + i * (BB * NHID); ap[i] = APK + (size_t)i * BB * NHID; }
  const size_t EW = 1024ull * 2048;   // packed elems per edge matrix

  for (int t = 0; t < TT; ++t) {
    const float* hprev = (t == 0) ? h0 : hbuf;
    const int* tok = tokens + t * BB;

    pack_x<<<(BB * NHID) / 256, 256, 0, stream>>>(encW, tok, xhPk);

    PhaseArgs p0{}; // init: s0 = hprev + sig(c0)*(tanh(h0c)-hprev), K=2048
    p0.W0a = W0p; p0.pa0 = xhPk; p0.s0a = hprev; p0.d0a = s[0]; p0.pd0 = ap[0];
    p0.act0 = 2; p0.ktiles = 64;
    rnn_gemm<<<dim3(16, SPLITK, 1), 256, 0, stream>>>(p0, part);
    rnn_epi <<<dim3(128, 1, 1),     256, 0, stream>>>(p0, part);

    PhaseArgs p1{}; // edge0: s0 -> s1, sigmoid
    p1.W0a = Wsp; p1.pa0 = ap[0]; p1.s0a = s[0]; p1.d0a = s[1]; p1.pd0 = ap[1];
    p1.act0 = 0; p1.ktiles = 32;
    rnn_gemm<<<dim3(16, SPLITK, 1), 256, 0, stream>>>(p1, part);
    rnn_epi <<<dim3(128, 1, 1),     256, 0, stream>>>(p1, part);

    PhaseArgs p2{}; // edges 1,2,3: s1 -> s2(relu), s3(relu), s4(identity)
    p2.W0a = Wsp + 1 * EW; p2.pa0 = ap[1]; p2.s0a = s[1]; p2.d0a = s[2]; p2.pd0 = ap[2]; p2.act0 = 1;
    p2.W1a = Wsp + 2 * EW; p2.pa1 = ap[1]; p2.s1a = s[1]; p2.d1a = s[3]; p2.pd1 = ap[3]; p2.act1 = 1;
    p2.W2a = Wsp + 3 * EW; p2.pa2 = ap[1]; p2.s2a = s[1]; p2.d2a = s[4]; p2.pd2 = ap[4]; p2.act2 = 3;
    p2.ktiles = 32;
    rnn_gemm<<<dim3(16, SPLITK, 3), 256, 0, stream>>>(p2, part);
    rnn_epi <<<dim3(128, 1, 3),     256, 0, stream>>>(p2, part);

    PhaseArgs p3{}; // edge4: s2 -> s5 (tanh); edge6: s3 -> s7 (tanh)
    p3.W0a = Wsp + 4 * EW; p3.pa0 = ap[2]; p3.s0a = s[2]; p3.d0a = s[5]; p3.pd0 = ap[5]; p3.act0 = 2;
    p3.W1a = Wsp + 6 * EW; p3.pa1 = ap[3]; p3.s1a = s[3]; p3.d1a = s[7]; p3.pd1 = ap[7]; p3.act1 = 2;
    p3.ktiles = 32;
    rnn_gemm<<<dim3(16, SPLITK, 2), 256, 0, stream>>>(p3, part);
    rnn_epi <<<dim3(128, 1, 2),     256, 0, stream>>>(p3, part);

    PhaseArgs p4{}; // edge5: s5 -> s6 (sigmoid); edge7: s5 -> s8 (relu)
    p4.W0a = Wsp + 5 * EW; p4.pa0 = ap[5]; p4.s0a = s[5]; p4.d0a = s[6]; p4.pd0 = ap[6]; p4.act0 = 0;
    p4.W1a = Wsp + 7 * EW; p4.pa1 = ap[5]; p4.s1a = s[5]; p4.d1a = s[8]; p4.pd1 = ap[8]; p4.act1 = 1;
    p4.ktiles = 32;
    rnn_gemm<<<dim3(16, SPLITK, 2), 256, 0, stream>>>(p4, part);
    rnn_epi <<<dim3(128, 1, 2),     256, 0, stream>>>(p4, part);

    finalize_h<<<dim3((BB * NHID) / 256), 256, 0, stream>>>(
        S, hbuf, xhPk, hidPk, t);
  }

  dec_gemm  <<<dim3(625, 64), 128, 0, stream>>>(hidPk, Dp, out);
  logsoftmax<<<dim3(TT * BB), 256, 0, stream>>>(out, decb);
  hipMemcpyAsync(out + (size_t)TT * BB * NTOK, hbuf,
                 (size_t)BB * NHID * sizeof(float),
                 hipMemcpyDeviceToDevice, stream);
  (void)in_sizes; (void)n_in; (void)out_size; (void)ws_size;
}
